// unit_gcn_627065225272
// MI455X (gfx1250) — matrix operand order repaired and checked by reading, not compiled, not run on hardware
//
#include <hip/hip_runtime.h>
#include <hip/hip_bf16.h>
#include <math.h>

// ---------------------------------------------------------------------------
// unit GCN (DSTA-style) for MI455X / gfx1250, wave32 + WMMA f16->f32.
// Channel GEMMs (the FLOP-dominant part) run on v_wmma_f32_16x16x32_f16.
// ---------------------------------------------------------------------------

typedef __attribute__((ext_vector_type(16))) _Float16 v16h;
typedef __attribute__((ext_vector_type(8)))  float    v8f;

#define NN 48
#define CC 64
#define TT 300
#define VV 25
#define OO 64
#define II 16
#define DD 32
#define KK 3

#define TFRAMES 16                 // frames per block tile
#define PTILE   (TFRAMES * VV)     // 400 positions (== 25 tiles of 16)
#define NTILES  (PTILE / 16)       // 25
#define TTILES  ((TT + TFRAMES - 1) / TFRAMES)  // 19

// workspace layout (float offsets); total ~11.65M floats (~46.6 MB)
#define OFF_AN   0            // K*V*V  = 1875   normalized joint adjacency
#define OFF_RAN  1875         // K*V    = 75     row sums of An
#define OFF_GPN  1950         // V*V    = 625    normalized part graph
#define OFF_RGP  2575         // V      = 25
#define OFF_GBN  2600         // V*V    = 625    normalized body graph
#define OFF_RGB  3225         // V      = 25
#define OFF_PE   3250         // C*V    = 1600   positional encoding
#define OFF_XS   4850         // N*C*V  = 76800  sum over T of x
#define OFF_ATTV 81650        // N*K*O  = 9216   SK branch attention
#define OFF_ATT  90866        // N*V*V  = 30000  non-local attention
#define OFF_TH   120866       // N*I*T*V = 5760000  theta
#define OFF_PH   5880866      // N*I*T*V = 5760000  phi

// --- WMMA fragment loaders (layouts per CDNA5 ISA 7.12.2, wave32) ----------
// A (16x32 f16): lane L holds row m=L%16; halves h<8 -> k=h+8*(L/16),
//                halves h>=8 -> k=h+8+8*(L/16).
__device__ __forceinline__ v16h frag_a(const _Float16* W, int ldw, int m0, int k0) {
  int lane = threadIdx.x & 31;
  const _Float16* p = W + (m0 + (lane & 15)) * ldw + k0 + 8 * (lane >> 4);
  v16h a;
#pragma unroll
  for (int h = 0; h < 8; ++h) a[h] = p[h];
#pragma unroll
  for (int h = 0; h < 8; ++h) a[h + 8] = p[16 + h];
  return a;
}
// B (32x16 f16): lane L holds col n=L%16; halves h<8 -> k=h+8*(L/16),
// halves h>=8 -> k=h+8+8*(L/16). Stored [n][k] row-major in LDS.
__device__ __forceinline__ v16h frag_b(const _Float16* B, int ldb, int k0, int n0) {
  int lane = threadIdx.x & 31;
  const _Float16* p = B + (n0 + (lane & 15)) * ldb + k0 + 8 * (lane >> 4);
  v16h b;
#pragma unroll
  for (int h = 0; h < 8; ++h) b[h] = p[h];
#pragma unroll
  for (int h = 0; h < 8; ++h) b[h + 8] = p[16 + h];
  return b;
}

// ---------------------------------------------------------------------------
// Kernel A: normalize graphs, row sums, positional encoding (tiny, 1 block)
// ---------------------------------------------------------------------------
__global__ void k_setup(const float* __restrict__ A_fix, const float* __restrict__ PA,
                        const float* __restrict__ Gp, const float* __restrict__ Gb,
                        float* __restrict__ ws) {
  __shared__ float cs[KK * VV];
  __shared__ float cp[VV], cbo[VV];
  int tid = threadIdx.x;
  if (tid < KK * VV) {                       // column sums of A_fix+PA (sum over v)
    int k = tid / VV, w = tid % VV;
    float s = 0.f;
    for (int v = 0; v < VV; ++v) s += A_fix[(k * VV + v) * VV + w] + PA[(k * VV + v) * VV + w];
    cs[tid] = s + 0.001f;
  } else if (tid < KK * VV + VV) {
    int w = tid - KK * VV; float s = 0.f;
    for (int v = 0; v < VV; ++v) s += Gp[v * VV + w];
    cp[w] = s + 0.001f;
  } else if (tid < KK * VV + 2 * VV) {
    int w = tid - KK * VV - VV; float s = 0.f;
    for (int v = 0; v < VV; ++v) s += Gb[v * VV + w];
    cbo[w] = s + 0.001f;
  }
  __syncthreads();
  for (int i = tid; i < KK * VV * VV; i += blockDim.x) {
    int k = i / (VV * VV), w = i % VV;
    ws[OFF_AN + i] = (A_fix[i] + PA[i]) / cs[k * VV + w];
  }
  for (int i = tid; i < VV * VV; i += blockDim.x) {
    ws[OFF_GPN + i] = Gp[i] / cp[i % VV];
    ws[OFF_GBN + i] = Gb[i] / cbo[i % VV];
  }
  for (int i = tid; i < CC * VV; i += blockDim.x) {   // sinusoidal PE
    int c = i / VV, v = i % VV;
    float d = expf(-(float)(2 * (c >> 1)) * logf(10000.f) / (float)CC);
    float ang = d * (float)v;
    ws[OFF_PE + i] = (c & 1) ? cosf(ang) : sinf(ang);
  }
  __syncthreads();
  for (int i = tid; i < KK * VV; i += blockDim.x) {   // row sums of An
    int k = i / VV, v = i % VV; float s = 0.f;
    for (int w = 0; w < VV; ++w) s += ws[OFF_AN + (k * VV + v) * VV + w];
    ws[OFF_RAN + i] = s;
  }
  for (int i = tid; i < VV; i += blockDim.x) {
    float s1 = 0.f, s2 = 0.f;
    for (int w = 0; w < VV; ++w) { s1 += ws[OFF_GPN + i * VV + w]; s2 += ws[OFF_GBN + i * VV + w]; }
    ws[OFF_RGP + i] = s1; ws[OFF_RGB + i] = s2;
  }
}

// ---------------------------------------------------------------------------
// Kernel B: xs[n,c,v] = sum_t x[n,c,t,v]   (LDS ds_add_f32 reduction)
// ---------------------------------------------------------------------------
__global__ void k_xsum(const float* __restrict__ x, float* __restrict__ ws) {
  __shared__ float sv[VV];
  int nc = blockIdx.x;
  int tid = threadIdx.x;
  if (tid < VV) sv[tid] = 0.f;
  __syncthreads();
  const float* xp = x + (size_t)nc * (TT * VV);
  for (int i = tid; i < TT * VV; i += blockDim.x) atomicAdd(&sv[i % VV], xp[i]);
  __syncthreads();
  if (tid < VV) ws[OFF_XS + nc * VV + tid] = sv[tid];
}

// ---------------------------------------------------------------------------
// Kernel C: SK branch attention att_v[n,k,o] from global pooled branch sums
// ---------------------------------------------------------------------------
__global__ void k_skatt(float* __restrict__ ws,
                        const float* __restrict__ Wj, const float* __restrict__ bj,
                        const float* __restrict__ Wp, const float* __restrict__ bp,
                        const float* __restrict__ Wb, const float* __restrict__ bb,
                        const float* __restrict__ Wfc, const float* __restrict__ bfc,
                        const float* __restrict__ Wbr, const float* __restrict__ bbr) {
  __shared__ float tJ[KK][CC], tP[CC], tB[CC], glo[OO], zz[DD];
  int n = blockIdx.x, tid = threadIdx.x;   // blockDim = 64
  {
    int c = tid;
    const float* xs = ws + OFF_XS + (n * CC + c) * VV;
    for (int k = 0; k < KK; ++k) {
      float s = 0.f;
      for (int v = 0; v < VV; ++v) s += xs[v] * ws[OFF_RAN + k * VV + v];
      tJ[k][c] = s;
    }
    float sp = 0.f, sb = 0.f;
    for (int v = 0; v < VV; ++v) { sp += xs[v] * ws[OFF_RGP + v]; sb += xs[v] * ws[OFF_RGB + v]; }
    tP[c] = sp; tB[c] = sb;
  }
  __syncthreads();
  {
    int o = tid; float g = 0.f;
    for (int c = 0; c < CC; ++c) {
      float s = 0.f;
      for (int k = 0; k < KK; ++k) s += Wj[(k * OO + o) * CC + c] * tJ[k][c];
      s += Wp[o * CC + c] * tP[c] + Wb[o * CC + c] * tB[c];
      g += s;
    }
    float bjs = bj[o] + bj[OO + o] + bj[2 * OO + o];
    glo[o] = g / (float)(TT * VV) + bjs + bp[o] + bb[o];
  }
  __syncthreads();
  if (tid < DD) {
    float s = bfc[tid];
    for (int o = 0; o < OO; ++o) s += glo[o] * Wfc[tid * OO + o];
    zz[tid] = s;
  }
  __syncthreads();
  {
    int o = tid;
    float l[KK];
    for (int k = 0; k < KK; ++k) {
      float s = bbr[k * OO + o];
      for (int d = 0; d < DD; ++d) s += zz[d] * Wbr[(k * OO + o) * DD + d];
      l[k] = s;
    }
    float m = fmaxf(l[0], fmaxf(l[1], l[2]));
    float e0 = expf(l[0] - m), e1 = expf(l[1] - m), e2 = expf(l[2] - m);
    float inv = 1.f / (e0 + e1 + e2);
    ws[OFF_ATTV + (n * KK + 0) * OO + o] = e0 * inv;
    ws[OFF_ATTV + (n * KK + 1) * OO + o] = e1 * inv;
    ws[OFF_ATTV + (n * KK + 2) * OO + o] = e2 * inv;
  }
}

// ---------------------------------------------------------------------------
// Kernel D: theta/phi = [W_theta;W_phi] @ (x + pe)  via WMMA (M=16, K=64)
// ---------------------------------------------------------------------------
__global__ void k_thetaphi(const float* __restrict__ x,
                           const float* __restrict__ Wth, const float* __restrict__ bth,
                           const float* __restrict__ Wph, const float* __restrict__ bph,
                           float* __restrict__ ws) {
  __shared__ _Float16 s_xb[PTILE * 68];   // B operand: [pos][chan]
  __shared__ _Float16 s_W[32 * 68];       // rows 0-15 theta, 16-31 phi
  int n = blockIdx.y, tt = blockIdx.x, tid = threadIdx.x;
  int t0 = tt * TFRAMES;
  for (int i = tid; i < 32 * CC; i += 256) {
    int m = i >> 6, c = i & 63;
    float w = (m < 16) ? Wth[m * CC + c] : Wph[(m - 16) * CC + c];
    s_W[m * 68 + c] = (_Float16)w;
  }
  for (int pair = tid; pair < CC * TFRAMES; pair += 256) {
    int c = pair & 63, tf = pair >> 6, t = t0 + tf;
    const float* pe = ws + OFF_PE + c * VV;
    if (t < TT) {
      const float* xp = x + ((size_t)(n * CC + c) * TT + t) * VV;
      for (int v = 0; v < VV; ++v) s_xb[(tf * VV + v) * 68 + c] = (_Float16)(xp[v] + pe[v]);
    } else {
      for (int v = 0; v < VV; ++v) s_xb[(tf * VV + v) * 68 + c] = (_Float16)0.f;
    }
  }
  __syncthreads();

  int wv = tid >> 5;
  int mt = wv & 1;        // 0: theta, 1: phi
  int grp = wv >> 1;      // 0..3 -> position-tile groups of 7
  const v8f vzero = {0.f, 0.f, 0.f, 0.f, 0.f, 0.f, 0.f, 0.f};
  v8f acc[7];
#pragma unroll
  for (int j = 0; j < 7; ++j) acc[j] = vzero;

#pragma unroll
  for (int kc = 0; kc < 2; ++kc) {
    v16h a = frag_a(s_W, 68, mt * 16, kc * 32);
#pragma unroll
    for (int j = 0; j < 7; ++j) {
      int nt = grp * 7 + j; if (nt > 24) nt = 24;   // clamp (dup compute, no store)
      v16h b = frag_b(s_xb, 68, kc * 32, nt * 16);
      acc[j] = __builtin_amdgcn_wmma_f32_16x16x32_f16(false, a, false, b,
                                                      (short)0, acc[j], false, false);
    }
  }
  int lane = tid & 31, ncol = lane & 15, hh = lane >> 4;
  float* outb = ws + (mt == 0 ? OFF_TH : OFF_PH) + (size_t)n * (II * TT * VV);
  const float* bias = (mt == 0) ? bth : bph;
#pragma unroll
  for (int j = 0; j < 7; ++j) {
    int nt = grp * 7 + j; if (nt > 24) continue;
    int p = nt * 16 + ncol, tf = p / VV, pv = p % VV, t = t0 + tf;
    if (t >= TT) continue;
#pragma unroll
    for (int r = 0; r < 8; ++r) {
      int i = r + 8 * hh;
      outb[(size_t)i * (TT * VV) + t * VV + pv] = acc[j][r] + bias[i];
    }
  }
}

// ---------------------------------------------------------------------------
// Kernel E: non-local attention: softmax_v( (th^T ph)/(I*T) ), per sample
// ---------------------------------------------------------------------------
__global__ void k_att(float* __restrict__ ws) {
  __shared__ float thc[16 * VV], phc[16 * VV], sl[VV * VV];
  int n = blockIdx.x, tid = threadIdx.x;   // blockDim = 640
  const float* th = ws + OFF_TH + (size_t)n * (II * TT * VV);
  const float* ph = ws + OFF_PH + (size_t)n * (II * TT * VV);
  float acc = 0.f;
  int v = tid / VV, w = tid % VV;
  bool act = tid < VV * VV;
  for (int q0 = 0; q0 < II * TT; q0 += 16) {     // q = i*T + t (contiguous x25)
    for (int i = tid; i < 2 * 16 * VV; i += blockDim.x) {
      if (i < 16 * VV) thc[i] = th[q0 * VV + i];
      else             phc[i - 16 * VV] = ph[q0 * VV + (i - 16 * VV)];
    }
    __syncthreads();
    if (act) {
#pragma unroll
      for (int qq = 0; qq < 16; ++qq) acc += thc[qq * VV + v] * phc[qq * VV + w];
    }
    __syncthreads();
  }
  if (act) sl[tid] = acc / (float)(II * TT);
  __syncthreads();
  if (tid < VV) {                                 // softmax over v for column w
    int w2 = tid;
    float m = -1e30f;
    for (int v2 = 0; v2 < VV; ++v2) m = fmaxf(m, sl[v2 * VV + w2]);
    float s = 0.f;
    for (int v2 = 0; v2 < VV; ++v2) s += expf(sl[v2 * VV + w2] - m);
    float inv = 1.f / s;
    for (int v2 = 0; v2 < VV; ++v2)
      ws[OFF_ATT + n * VV * VV + v2 * VV + w2] = expf(sl[v2 * VV + w2] - m) * inv;
  }
}

// ---------------------------------------------------------------------------
// Kernel F: fused main pass. 6 graph-matrix x channel-GEMM products
// accumulated in WMMA tiles, SK attention folded into weights, then
// BN + residual + ReLU epilogue.
// ---------------------------------------------------------------------------
__global__ void k_main(const float* __restrict__ x,
                       const float* __restrict__ Wj, const float* __restrict__ bj,
                       const float* __restrict__ Wnl, const float* __restrict__ bnl,
                       const float* __restrict__ Wp, const float* __restrict__ bp,
                       const float* __restrict__ Wb, const float* __restrict__ bb,
                       const float* __restrict__ gamma, const float* __restrict__ beta,
                       const float* __restrict__ mean, const float* __restrict__ var,
                       const float* __restrict__ ws, float* __restrict__ out) {
  __shared__ _Float16 s_xb[PTILE * 68];   // 54.4 KB  B operand [pos][chan]
  __shared__ _Float16 s_wA[OO * 68];      //  8.7 KB  scaled weights (A operand)
  __shared__ float    s_Mg[VV * VV];      //  graph matrix
  __shared__ float    s_av[KK * OO];      //  branch attention for this n
  __shared__ float    s_cb[OO], s_sc[OO], s_sh[OO];

  int n = blockIdx.y, tt = blockIdx.x, tid = threadIdx.x;
  int t0 = tt * TFRAMES;

  if (tid < KK * OO) s_av[tid] = ws[OFF_ATTV + n * KK * OO + tid];
  if (tid >= 192 && tid < 256) {
    int o = tid - 192;
    float a0 = ws[OFF_ATTV + n * KK * OO + o];
    float a1 = ws[OFF_ATTV + n * KK * OO + OO + o];
    float a2 = ws[OFF_ATTV + n * KK * OO + 2 * OO + o];
    s_cb[o] = a0 * (bj[o] + bj[OO + o] + bj[2 * OO + o]) + a1 * bp[o] + a2 * bb[o] + bnl[o];
    float inv = gamma[o] * rsqrtf(var[o] + 1e-5f);
    s_sc[o] = inv;
    s_sh[o] = beta[o] - mean[o] * inv;
  }
  // warm the x tile into cache (global_prefetch_b8)
  for (int i = tid; i < CC; i += 256)
    __builtin_prefetch(x + ((size_t)(n * CC + i) * TT + t0) * VV, 0, 1);
  __syncthreads();

  int wv = tid >> 5, lane = tid & 31;
  int mt = wv >> 1;           // O-tile 0..3
  int grp = wv & 1;           // position-tile halves (0..12 / 12..24)
  int ntb = grp ? 12 : 0;
  const v8f vzero = {0.f, 0.f, 0.f, 0.f, 0.f, 0.f, 0.f, 0.f};
  v8f acc[13];
#pragma unroll
  for (int j = 0; j < 13; ++j) acc[j] = vzero;

  for (int g = 0; g < 6; ++g) {
    const float* Msrc = (g < 3)  ? (ws + OFF_AN + g * VV * VV)
                      : (g == 3) ? (ws + OFF_GPN)
                      : (g == 4) ? (ws + OFF_GBN)
                                 : (ws + OFF_ATT + n * VV * VV);
    const float* Wsrc = (g < 3)  ? (Wj + g * OO * CC)
                      : (g == 3) ? Wp
                      : (g == 4) ? Wb : Wnl;
    const float* avs  = (g < 3)  ? s_av
                      : (g == 3) ? (s_av + OO)
                      : (g == 4) ? (s_av + 2 * OO) : nullptr;

    for (int i = tid; i < VV * VV; i += 256) s_Mg[i] = Msrc[i];
    for (int i = tid; i < OO * CC; i += 256) {
      int o = i >> 6, c = i & 63;
      float sc = avs ? avs[o] : 1.0f;                 // fold SK attention into W
      s_wA[o * 68 + c] = (_Float16)(Wsrc[i] * sc);
    }
    __syncthreads();

    // graph mult on VALU: B[p][c] = sum_v x[c,t,v] * Mg[v,pv]  (f32 -> f16)
    for (int pair = tid; pair < CC * TFRAMES; pair += 256) {
      int c = pair & 63, tf = pair >> 6, t = t0 + tf;
      float xr[VV];
      if (t < TT) {
        const float* xp = x + ((size_t)(n * CC + c) * TT + t) * VV;
#pragma unroll
        for (int v = 0; v < VV; ++v) xr[v] = xp[v];
      } else {
#pragma unroll
        for (int v = 0; v < VV; ++v) xr[v] = 0.f;
      }
      for (int pv = 0; pv < VV; ++pv) {
        float s = 0.f;
#pragma unroll
        for (int v = 0; v < VV; ++v) s += xr[v] * s_Mg[v * VV + pv];
        s_xb[(tf * VV + pv) * 68 + c] = (_Float16)s;
      }
    }
    __syncthreads();

    // WMMA: acc += W~ (16x64) @ B (64x16-per-tile), K split into 2x32
#pragma unroll
    for (int kc = 0; kc < 2; ++kc) {
      v16h a = frag_a(s_wA, 68, mt * 16, kc * 32);
#pragma unroll
      for (int j = 0; j < 13; ++j) {
        v16h b = frag_b(s_xb, 68, kc * 32, (ntb + j) * 16);
        acc[j] = __builtin_amdgcn_wmma_f32_16x16x32_f16(false, a, false, b,
                                                        (short)0, acc[j], false, false);
      }
    }
    __syncthreads();   // LDS reused next graph matrix
  }

  // epilogue: bias, BN (eval), residual, relu
  int ncol = lane & 15, hh = lane >> 4;
#pragma unroll
  for (int j = 0; j < 13; ++j) {
    if (grp == 1 && j == 0) continue;   // tile 12 stored by group 0
    int nt = ntb + j;
    int p = nt * 16 + ncol, tf = p / VV, pv = p % VV, t = t0 + tf;
    if (t >= TT) continue;
#pragma unroll
    for (int r = 0; r < 8; ++r) {
      int o = mt * 16 + r + 8 * hh;
      size_t idx = ((size_t)(n * OO + o) * TT + t) * VV + pv;
      float val = acc[j][r] + s_cb[o];
      val = val * s_sc[o] + s_sh[o];
      val += x[idx];
      out[idx] = fmaxf(val, 0.f);
    }
  }
}

// ---------------------------------------------------------------------------
extern "C" void kernel_launch(void* const* d_in, const int* in_sizes, int n_in,
                              void* d_out, int out_size, void* d_ws, size_t ws_size,
                              hipStream_t stream) {
  (void)in_sizes; (void)n_in; (void)out_size; (void)ws_size;
  const float* x       = (const float*)d_in[0];
  const float* PA      = (const float*)d_in[1];
  const float* A_fix   = (const float*)d_in[2];
  const float* G_part  = (const float*)d_in[3];
  const float* G_body  = (const float*)d_in[4];
  const float* W_joint = (const float*)d_in[5];
  const float* b_joint = (const float*)d_in[6];
  const float* W_theta = (const float*)d_in[7];
  const float* b_theta = (const float*)d_in[8];
  const float* W_phi   = (const float*)d_in[9];
  const float* b_phi   = (const float*)d_in[10];
  const float* W_nl    = (const float*)d_in[11];
  const float* b_nl    = (const float*)d_in[12];
  const float* W_part  = (const float*)d_in[13];
  const float* b_part  = (const float*)d_in[14];
  const float* W_body  = (const float*)d_in[15];
  const float* b_body  = (const float*)d_in[16];
  const float* W_fc    = (const float*)d_in[17];
  const float* b_fc    = (const float*)d_in[18];
  const float* W_br    = (const float*)d_in[19];
  const float* b_br    = (const float*)d_in[20];
  const float* bn_g    = (const float*)d_in[21];
  const float* bn_b    = (const float*)d_in[22];
  const float* bn_m    = (const float*)d_in[23];
  const float* bn_v    = (const float*)d_in[24];
  float* ws  = (float*)d_ws;    // needs ~46.6 MB
  float* out = (float*)d_out;

  k_setup<<<1, 256, 0, stream>>>(A_fix, PA, G_part, G_body, ws);
  k_xsum<<<NN * CC, 128, 0, stream>>>(x, ws);
  k_skatt<<<NN, 64, 0, stream>>>(ws, W_joint, b_joint, W_part, b_part,
                                 W_body, b_body, W_fc, b_fc, W_br, b_br);
  k_thetaphi<<<dim3(TTILES, NN), 256, 0, stream>>>(x, W_theta, b_theta, W_phi, b_phi, ws);
  k_att<<<NN, 640, 0, stream>>>(ws);
  k_main<<<dim3(TTILES, NN), 256, 0, stream>>>(x, W_joint, b_joint, W_nl, b_nl,
                                               W_part, b_part, W_body, b_body,
                                               bn_g, bn_b, bn_m, bn_v, ws, out);
}
